// Model_49555332661574
// MI455X (gfx1250) — matrix operand order repaired and checked by reading, not compiled, not run on hardware
//
#include <hip/hip_runtime.h>
#include <hip/hip_bf16.h>
#include <math.h>

typedef __attribute__((ext_vector_type(16))) _Float16 v16h;
typedef __attribute__((ext_vector_type(8)))  _Float16 v8h;
typedef __attribute__((ext_vector_type(8)))  float    v8f;

#define BS    16
#define LMEM  1024
#define TGT   256
#define HID   512
#define NKV   1024          // hidden*sub = Wk output width
#define TVOC  2000
#define VOC   3000
#define SRC   2048          // l*sub
#define SKEY  2049          // SRC + sentinel
#define NEGC  (-1000000000.0f)
#define EPSC  (1.1920929e-07f)
#define FMINC (-3.4028234663852886e38f)

// ---------------- elementwise f32 -> f16 ----------------
__global__ void k_f32_to_f16(const float* __restrict__ in,
                             _Float16* __restrict__ out, long n) {
  long i = (long)blockIdx.x * blockDim.x + threadIdx.x;
  if (i < n) out[i] = (_Float16)in[i];
}

// f32 (K x N row-major) -> f16 transposed (N x K row-major)
__global__ void k_f32_to_f16_t(const float* __restrict__ in,
                               _Float16* __restrict__ out, int K, int N) {
  long i = (long)blockIdx.x * blockDim.x + threadIdx.x;
  if (i < (long)K * N) {
    int k = (int)(i / N), n = (int)(i % N);
    out[(size_t)n * K + k] = (_Float16)in[i];
  }
}

// sentinel (1,512) f32 -> f16 row 2048 of each batch's key buffer
__global__ void k_sentinel(const float* __restrict__ sent,
                           _Float16* __restrict__ Kf16) {
  int b = blockIdx.x;
  for (int h = threadIdx.x; h < HID; h += blockDim.x)
    Kf16[(size_t)b * SKEY * HID + (size_t)SRC * HID + h] = (_Float16)sent[h];
}

// ---- per-lane WMMA operand loads (both = 2x global_load_b128) --------------
// A 16x32 f16 layout: lane holds K runs [8*sel, +8) and [16+8*sel, +8)
__device__ __forceinline__ v16h load_a16(const _Float16* __restrict__ Arow,
                                         int k0, int sel) {
  v16h av;
  *reinterpret_cast<v8h*>(&av) =
      *reinterpret_cast<const v8h*>(Arow + k0 + 8 * sel);
  *(reinterpret_cast<v8h*>(&av) + 1) =
      *reinterpret_cast<const v8h*>(Arow + k0 + 16 + 8 * sel);
  return av;
}
// B 32x16 f16 layout with B^T stored N-major: lane holds K runs [8*sel, +8) and [16+8*sel, +8)
__device__ __forceinline__ v16h load_bt16(const _Float16* __restrict__ Brow,
                                          int k0, int sel) {
  v16h bv;
  *reinterpret_cast<v8h*>(&bv) =
      *reinterpret_cast<const v8h*>(Brow + k0 + 8 * sel);
  *(reinterpret_cast<v8h*>(&bv) + 1) =
      *reinterpret_cast<const v8h*>(Brow + k0 + 16 + 8 * sel);
  return bv;
}

__device__ __forceinline__ v8f wmma16(v16h a, v16h b, v8f c) {
  return __builtin_amdgcn_wmma_f32_16x16x32_f16(false, a, false, b,
                                                (short)0, c, false, false);
}

// epilogue for one 16x16 C tile: bias + optional exact gelu, batch-rerouted
__device__ __forceinline__ void store_tile(
    v8f c, _Float16* __restrict__ out, const float* __restrict__ bias,
    int rbase, int cbase, int lm, int sel, int N, int rows_per_batch,
    long batch_out_stride, int act) {
  const int n = cbase + lm;
  const float bb = bias[n];
#pragma unroll
  for (int r = 0; r < 8; ++r) {            // C/D: M = r + 8*half, N = lane&15
    int m = rbase + r + 8 * sel;
    float v = c[r] + bb;
    if (act) v = 0.5f * v * (1.0f + erff(v * 0.70710678118654752f));
    size_t o = (size_t)(m / rows_per_batch) * batch_out_stride
             + (size_t)(m % rows_per_batch) * N + n;
    out[o] = (_Float16)v;
  }
}

// ---------------- WMMA GEMM: out_f16 = A(MxK) * Bt(NxK)^T + bias ------------
// 8 waves/block, each wave computes a 32x32 C block -> block tile 64x128.
__global__ __launch_bounds__(256) void k_wmma_gemm(
    const _Float16* __restrict__ A, const _Float16* __restrict__ Bt,
    const float* __restrict__ bias, _Float16* __restrict__ out,
    int M, int N, int K, int rows_per_batch, long batch_out_stride, int act) {
  const int lane = threadIdx.x & 31;
  const int wid  = threadIdx.x >> 5;
  const int wm = wid >> 2, wn = wid & 3;          // 2x4 wave grid
  const int row0 = blockIdx.y * 64 + wm * 32;
  const int col0 = blockIdx.x * 128 + wn * 32;
  const int lm  = lane & 15;
  const int sel = lane >> 4;

  const _Float16* Ar0 = A  + (size_t)(row0 + lm) * K;
  const _Float16* Ar1 = A  + (size_t)(row0 + 16 + lm) * K;
  const _Float16* Br0 = Bt + (size_t)(col0 + lm) * K;
  const _Float16* Br1 = Bt + (size_t)(col0 + 16 + lm) * K;

  v8f c00 = {}, c01 = {}, c10 = {}, c11 = {};
  for (int k0 = 0; k0 < K; k0 += 32) {
    v16h a0 = load_a16(Ar0, k0, sel);
    v16h a1 = load_a16(Ar1, k0, sel);
    v16h b0 = load_bt16(Br0, k0, sel);
    v16h b1 = load_bt16(Br1, k0, sel);
    if (k0 + 32 < K) {
      __builtin_prefetch(Ar0 + k0 + 32, 0, 0);    // global_prefetch_b8
      __builtin_prefetch(Br0 + k0 + 32, 0, 0);
    }
    c00 = wmma16(a0, b0, c00);
    c01 = wmma16(a0, b1, c01);
    c10 = wmma16(a1, b0, c10);
    c11 = wmma16(a1, b1, c11);
  }

  store_tile(c00, out, bias, row0,      col0,      lm, sel, N, rows_per_batch, batch_out_stride, act);
  store_tile(c01, out, bias, row0,      col0 + 16, lm, sel, N, rows_per_batch, batch_out_stride, act);
  store_tile(c10, out, bias, row0 + 16, col0,      lm, sel, N, rows_per_batch, batch_out_stride, act);
  store_tile(c11, out, bias, row0 + 16, col0 + 16, lm, sel, N, rows_per_batch, batch_out_stride, act);
}

// ---------------- batched attention scores: Q(256x512) x K(2049x512)^T ------
// K matrix is naturally N-major (keys row-major over hidden) -> load_bt16.
__global__ __launch_bounds__(256) void k_wmma_attn(
    const _Float16* __restrict__ Q, const _Float16* __restrict__ Kmat,
    float* __restrict__ scores, float scale) {
  const int lane = threadIdx.x & 31;
  const int wid  = threadIdx.x >> 5;
  const int wm = wid >> 2, wn = wid & 3;
  const int b    = blockIdx.z;
  const int row0 = blockIdx.y * 64 + wm * 32;
  const int col0 = blockIdx.x * 128 + wn * 32;
  const int lm  = lane & 15;
  const int sel = lane >> 4;

  const int n0 = col0 + lm, n1 = col0 + 16 + lm;
  const bool ok0 = (n0 < SKEY), ok1 = (n1 < SKEY);

  const _Float16* Qb  = Q + (size_t)b * TGT * HID;
  const _Float16* Kb  = Kmat + (size_t)b * SKEY * HID;
  const _Float16* Ar0 = Qb + (size_t)(row0 + lm) * HID;
  const _Float16* Ar1 = Qb + (size_t)(row0 + 16 + lm) * HID;
  const _Float16* Br0 = Kb + (size_t)(ok0 ? n0 : 0) * HID;
  const _Float16* Br1 = Kb + (size_t)(ok1 ? n1 : 0) * HID;
  const v16h zerov = {};

  v8f c00 = {}, c01 = {}, c10 = {}, c11 = {};
  for (int k0 = 0; k0 < HID; k0 += 32) {
    v16h a0 = load_a16(Ar0, k0, sel);
    v16h a1 = load_a16(Ar1, k0, sel);
    v16h b0 = ok0 ? load_bt16(Br0, k0, sel) : zerov;
    v16h b1 = ok1 ? load_bt16(Br1, k0, sel) : zerov;
    if (k0 + 32 < HID) __builtin_prefetch(Ar0 + k0 + 32, 0, 0);
    c00 = wmma16(a0, b0, c00);
    c01 = wmma16(a0, b1, c01);
    c10 = wmma16(a1, b0, c10);
    c11 = wmma16(a1, b1, c11);
  }

  float* sb = scores + (size_t)b * TGT * SKEY;
#pragma unroll
  for (int r = 0; r < 8; ++r) {
    int m0 = row0 + r + 8 * sel, m1 = m0 + 16;
    if (ok0) {
      sb[(size_t)m0 * SKEY + n0] = c00[r] * scale;
      sb[(size_t)m1 * SKEY + n0] = c10[r] * scale;
    }
    if (ok1) {
      sb[(size_t)m0 * SKEY + n1] = c01[r] * scale;
      sb[(size_t)m1 * SKEY + n1] = c11[r] * scale;
    }
  }
}

// ---------------- fused mask/log-softmax/gate/scatter/mixture ---------------
__global__ __launch_bounds__(256) void k_finalize(
    const float* __restrict__ scores, const int* __restrict__ content_e,
    const float* __restrict__ out_logits, float* __restrict__ out) {
  __shared__ float sh_att[SKEY];   // masked attention row
  __shared__ float sh_p[VOC];      // scattered pointer probs (12 KB of LDS)
  __shared__ float sh_ol[TVOC];    // out_logits row
  __shared__ float red[256];

  const int t = blockIdx.x, b = blockIdx.y, tid = threadIdx.x;
  const float* srow = scores + ((size_t)b * TGT + t) * SKEY;
  const int*   ce   = content_e + (size_t)b * SRC;

  for (int s = tid; s < SKEY; s += 256) {
    float v = srow[s];
    if (s < SRC && ce[s] == 0) v = NEGC;     // key padding mask (sentinel free)
    sh_att[s] = v;
  }
  for (int v = tid; v < VOC; v += 256)  sh_p[v] = 0.0f;
  for (int v = tid; v < TVOC; v += 256)
    sh_ol[v] = out_logits[((size_t)b * TGT + t) * TVOC + v];
  __syncthreads();

  // log-softmax over 2049 keys
  float mx = -INFINITY;
  for (int s = tid; s < SKEY; s += 256) mx = fmaxf(mx, sh_att[s]);
  red[tid] = mx; __syncthreads();
  for (int o = 128; o > 0; o >>= 1) {
    if (tid < o) red[tid] = fmaxf(red[tid], red[tid + o]);
    __syncthreads();
  }
  mx = red[0]; __syncthreads();
  float sm = 0.0f;
  for (int s = tid; s < SKEY; s += 256) sm += expf(sh_att[s] - mx);
  red[tid] = sm; __syncthreads();
  for (int o = 128; o > 0; o >>= 1) {
    if (tid < o) red[tid] += red[tid + o];
    __syncthreads();
  }
  const float lse = mx + logf(red[0]); __syncthreads();
  const float gate = sh_att[SRC] - lse;          // log prob of sentinel

  // scatter exp(log-softmax) into vocab table (per-batch segment ids)
  for (int s = tid; s < SRC; s += 256)
    atomicAdd(&sh_p[ce[s]], expf(sh_att[s] - lse));

  // log-softmax of out_logits row (2000)
  float mo = -INFINITY;
  for (int v = tid; v < TVOC; v += 256) mo = fmaxf(mo, sh_ol[v]);
  red[tid] = mo; __syncthreads();                // also fences the scatter
  for (int o = 128; o > 0; o >>= 1) {
    if (tid < o) red[tid] = fmaxf(red[tid], red[tid + o]);
    __syncthreads();
  }
  mo = red[0]; __syncthreads();
  float so = 0.0f;
  for (int v = tid; v < TVOC; v += 256) so += expf(sh_ol[v] - mo);
  red[tid] = so; __syncthreads();
  for (int o = 128; o > 0; o >>= 1) {
    if (tid < o) red[tid] += red[tid + o];
    __syncthreads();
  }
  const float lse_o = mo + logf(red[0]); __syncthreads();

  const float eg = expf(gate);
  const float d1 = log1pf(-eg + EPSC);
  const float d2 = logf(1.0f - eg + EPSC);

  float* orow = out + ((size_t)b * TGT + t) * VOC;
  for (int v = tid; v < VOC; v += 256) {
    float t2 = logf(sh_p[v] + EPSC) - d1;
    if (isinf(t2) && t2 < 0.0f) t2 = FMINC;      // isneginf -> float32 min
    t2 += d2;
    float t1 = (v < TVOC) ? (sh_ol[v] - lse_o + gate) : -INFINITY;
    float m = fmaxf(t1, t2);
    orow[v] = (m == -INFINITY) ? -INFINITY
                               : m + logf(expf(t1 - m) + expf(t2 - m));
  }
}

extern "C" void kernel_launch(void* const* d_in, const int* in_sizes, int n_in,
                              void* d_out, int out_size, void* d_ws,
                              size_t ws_size, hipStream_t stream) {
  const float* out_logits = (const float*)d_in[0];
  const float* feature    = (const float*)d_in[1];
  const float* memory_raw = (const float*)d_in[2];
  const int*   content_e  = (const int*)d_in[3];
  const float* Wq         = (const float*)d_in[4];
  const float* bq         = (const float*)d_in[5];
  const float* Wk         = (const float*)d_in[6];
  const float* bk         = (const float*)d_in[7];
  const float* sentinel   = (const float*)d_in[8];
  float* out = (float*)d_out;

  char* ws = (char*)d_ws;
  size_t off = 0;
  auto alloc = [&](size_t bytes) -> void* {
    void* p = ws + off;
    off = (off + bytes + 255) & ~(size_t)255;
    return p;
  };
  const long n_mem  = (long)BS * LMEM * HID;   // 8,388,608
  const long n_wk   = (long)HID * NKV;         //   524,288
  const long n_wq   = (long)HID * HID;         //   262,144
  const long n_feat = (long)BS * TGT * HID;    // 2,097,152
  _Float16* mem_f16  = (_Float16*)alloc(n_mem  * 2);
  _Float16* wkT_f16  = (_Float16*)alloc(n_wk   * 2);  // (NKV, HID) N-major
  _Float16* wqT_f16  = (_Float16*)alloc(n_wq   * 2);  // (HID, HID) N-major
  _Float16* feat_f16 = (_Float16*)alloc(n_feat * 2);
  _Float16* Kf16     = (_Float16*)alloc((size_t)BS * SKEY * HID * 2);
  _Float16* Qf16     = (_Float16*)alloc((size_t)BS * TGT * HID * 2);
  float*    scores   = (float*)alloc((size_t)BS * TGT * SKEY * 4);

  k_f32_to_f16<<<(unsigned)((n_mem  + 255) / 256), 256, 0, stream>>>(memory_raw, mem_f16, n_mem);
  k_f32_to_f16<<<(unsigned)((n_feat + 255) / 256), 256, 0, stream>>>(feature, feat_f16, n_feat);
  k_f32_to_f16_t<<<(unsigned)((n_wk + 255) / 256), 256, 0, stream>>>(Wk, wkT_f16, HID, NKV);
  k_f32_to_f16_t<<<(unsigned)((n_wq + 255) / 256), 256, 0, stream>>>(Wq, wqT_f16, HID, HID);
  k_sentinel<<<BS, 256, 0, stream>>>(sentinel, Kf16);

  // memory = memory_raw @ Wk + bk, written straight into (bs,2049,512) K layout
  k_wmma_gemm<<<dim3(NKV / 128, (BS * LMEM) / 64), 256, 0, stream>>>(
      mem_f16, wkT_f16, bk, Kf16,
      BS * LMEM, NKV, HID, LMEM, (long)SKEY * HID, /*act=*/0);

  // query = gelu(feature @ Wq + bq)
  k_wmma_gemm<<<dim3(HID / 128, (BS * TGT) / 64), 256, 0, stream>>>(
      feat_f16, wqT_f16, bq, Qf16,
      BS * TGT, HID, HID, BS * TGT, 0L, /*act=*/1);

  // atten = Q K^T / sqrt(hidden)
  k_wmma_attn<<<dim3((SKEY + 127) / 128, TGT / 64, BS), 256, 0, stream>>>(
      Qf16, Kf16, scores, 0.04419417382415922f);

  // mask + log-softmax + gate + scatter + mixture
  k_finalize<<<dim3(TGT, BS), 256, 0, stream>>>(scores, content_e, out_logits, out);
}
